// Model_33646773797444
// MI455X (gfx1250) — matrix operand order repaired and checked by reading, not compiled, not run on hardware
//
#include <hip/hip_runtime.h>
#include <hip/hip_bf16.h>

// ---------------------------------------------------------------------------
// MI455X (gfx1250, wave32) implementation of the reference attention block.
// All matmuls: v_wmma_f32_16x16x32_bf16 (bf16 in, f32 accumulate).
// Weights are pre-transposed so WMMA B-operands load contiguously from global
// (lane = column, 16 contiguous K elements) -> barrier-free GEMM main loops.
// Transpose kernels stage tiles with GLOBAL_LOAD_ASYNC_TO_LDS_B128 (ASYNCcnt).
// GEMM wave tile = 32(M) x 64(N): 2 A-frags x 4 B-frags -> 8 WMMAs per
// K-step of 32 with only 12 16B loads (operand reuse up 1.5x vs 16x128).
// ---------------------------------------------------------------------------

typedef __bf16 bf16;
typedef __attribute__((ext_vector_type(16))) __bf16 v16bf;
typedef __attribute__((ext_vector_type(8)))  __bf16 v8bf;
typedef __attribute__((ext_vector_type(8)))  float  v8f;

#define B_   2
#define S_   2048
#define H_   2048
#define NH_  16
#define HD_  128
#define TS_  512

#define USE_ASYNC_LDS 1

__device__ __forceinline__ v8f wmma_bf16(v16bf a, v16bf b, v8f c) {
  return __builtin_amdgcn_wmma_f32_16x16x32_bf16(false, a, false, b, (short)0, c,
                                                 false, false);
}

// global -> LDS 16-byte async copy (CDNA5 GLOBAL_LOAD_ASYNC_TO_LDS_B128).
// LDS byte address = low 32 bits of the generic pointer (aperture rule).
__device__ __forceinline__ void async_g2l16(void* lds, const void* g) {
#if USE_ASYNC_LDS
  unsigned l = (unsigned)(unsigned long long)lds;
  unsigned long long ga = (unsigned long long)g;
  asm volatile("global_load_async_to_lds_b128 %0, %1, off"
               :: "v"(l), "v"(ga) : "memory");
#else
  *reinterpret_cast<v8bf*>(lds) = *reinterpret_cast<const v8bf*>(g);
#endif
}
__device__ __forceinline__ void async_wait() {
#if USE_ASYNC_LDS
  asm volatile("s_wait_asynccnt 0x0" ::: "memory");
#endif
}

// ---------------------------------------------------------------------------
// fp32 -> bf16 (for hidden states, the WMMA A operand: row-major is native)
// ---------------------------------------------------------------------------
__global__ void cvt_f32_bf16(const float* __restrict__ in, bf16* __restrict__ out,
                             int n) {
  int i = blockIdx.x * blockDim.x + threadIdx.x;
  if (i < n) out[i] = (bf16)in[i];
}

// ---------------------------------------------------------------------------
// Weight transpose + convert: out[n*ldout + k] = (bf16) in[k*ldin + n].
// 32x32 tiles staged global->LDS via async b128, then written transposed.
// ---------------------------------------------------------------------------
__global__ __launch_bounds__(256)
void transpose_cvt(const float* __restrict__ in, int ldin,
                   bf16* __restrict__ out, int ldout) {
  __shared__ __align__(16) float tile[32 * 36];  // [k][n], stride 36 floats
  const int tid = threadIdx.x;
  const int k0 = blockIdx.y * 32;
  const int n0 = blockIdx.x * 32;
  {
    int r = tid >> 3, c = (tid & 7) * 4;  // 256 x 16B = 4KB tile
    async_g2l16(&tile[r * 36 + c], &in[(size_t)(k0 + r) * ldin + n0 + c]);
  }
  async_wait();
  __syncthreads();
  int ny = tid >> 3, kq = (tid & 7) * 4;
#pragma unroll
  for (int j = 0; j < 4; ++j)
    out[(size_t)(n0 + ny) * ldout + k0 + kq + j] = (bf16)tile[(kq + j) * 36 + ny];
}

// ---------------------------------------------------------------------------
// bf16 per-batch transpose (V -> vT[b][h*HD+d][s]).
// ---------------------------------------------------------------------------
__global__ __launch_bounds__(256)
void transpose_b16(const bf16* __restrict__ in, bf16* __restrict__ out) {
  __shared__ __align__(16) bf16 tile[32 * 40];  // [s][n], stride 40 bf16
  const int tid = threadIdx.x;
  const size_t sb = (size_t)blockIdx.z * S_ * H_;
  const int s0 = blockIdx.y * 32;
  const int n0 = blockIdx.x * 32;
  if (tid < 128) {
    int r = tid >> 2, c = (tid & 3) * 8;  // 128 x 16B = 2KB tile
    async_g2l16(&tile[r * 40 + c], &in[sb + (size_t)(s0 + r) * H_ + n0 + c]);
  }
  async_wait();
  __syncthreads();
  int ny = tid >> 3, kq = (tid & 7) * 4;
#pragma unroll
  for (int j = 0; j < 4; ++j)
    out[sb + (size_t)(n0 + ny) * S_ + s0 + kq + j] = tile[(kq + j) * 40 + ny];
}

// ---------------------------------------------------------------------------
// GEMM: C[M x N] = A[M x K](bf16, row-major) * Bt[N x K](bf16, transposed)
//       (+ Add bf16, row-major)
// Block = 256 threads = 8 waves in a 4(M) x 2(N) grid -> 128 x 128 tile.
// Wave = 32(M) x 64(N): 2 A-fragments x 4 B-fragments -> 8 WMMAs per K-step
// of 32 with 12 contiguous 16B global loads. No LDS, no barriers.
// ---------------------------------------------------------------------------
template <bool OUT_BF16, bool HAS_ADD>
__global__ __launch_bounds__(256)
void gemm_bf16t(const bf16* __restrict__ A, int lda,
                const bf16* __restrict__ Bt, int ldbt,
                const bf16* __restrict__ Add, int ldadd,
                float* __restrict__ Cf, bf16* __restrict__ Cb, int ldc, int K) {
  const int tid  = threadIdx.x;
  const int w    = tid >> 5;
  const int ln32 = tid & 31;
  const int half = ln32 >> 4;
  const int ln   = ln32 & 15;
  const int wm   = w & 3;
  const int wn   = w >> 2;
  const int m0   = blockIdx.y * 128 + wm * 32;
  const int n0   = blockIdx.x * 128 + wn * 64;

  const bf16* ap0 = &A[(size_t)(m0 + ln) * lda];
  const bf16* ap1 = &A[(size_t)(m0 + 16 + ln) * lda];
  const bf16* bp[4];
#pragma unroll
  for (int t = 0; t < 4; ++t)
    bp[t] = &Bt[(size_t)(n0 + t * 16 + ln) * ldbt + half * 16];

  v8f acc[2][4];
  v8f z = {};
#pragma unroll
  for (int mi = 0; mi < 2; ++mi)
#pragma unroll
    for (int t = 0; t < 4; ++t) acc[mi][t] = z;

  for (int k0 = 0; k0 < K; k0 += 32) {
    v16bf afr0, afr1;
    {
      v8bf a0 = *reinterpret_cast<const v8bf*>(ap0 + k0 + half * 16);
      v8bf a1 = *reinterpret_cast<const v8bf*>(ap0 + k0 + 8 + half * 16);
      v8bf a2 = *reinterpret_cast<const v8bf*>(ap1 + k0 + half * 16);
      v8bf a3 = *reinterpret_cast<const v8bf*>(ap1 + k0 + 8 + half * 16);
#pragma unroll
      for (int i = 0; i < 8; ++i) {
        afr0[i] = a0[i]; afr0[8 + i] = a1[i];
        afr1[i] = a2[i]; afr1[8 + i] = a3[i];
      }
    }
#pragma unroll
    for (int t = 0; t < 4; ++t) {
      v8bf b0 = *reinterpret_cast<const v8bf*>(bp[t] + k0);
      v8bf b1 = *reinterpret_cast<const v8bf*>(bp[t] + k0 + 8);
      v16bf bfr;
#pragma unroll
      for (int i = 0; i < 8; ++i) { bfr[i] = b0[i]; bfr[8 + i] = b1[i]; }
      acc[0][t] = wmma_bf16(afr0, bfr, acc[0][t]);
      acc[1][t] = wmma_bf16(afr1, bfr, acc[1][t]);
    }
  }

#pragma unroll
  for (int mi = 0; mi < 2; ++mi)
#pragma unroll
    for (int t = 0; t < 4; ++t) {
      const int n_g = n0 + t * 16 + ln;
#pragma unroll
      for (int r = 0; r < 8; ++r) {
        int m_g = m0 + mi * 16 + r + half * 8;
        float v = acc[mi][t][r];
        if (HAS_ADD) v += (float)Add[(size_t)m_g * ldadd + n_g];
        if (OUT_BF16) Cb[(size_t)m_g * ldc + n_g] = (bf16)v;
        else          Cf[(size_t)m_g * ldc + n_g] = v;
      }
    }
}

// ---------------------------------------------------------------------------
// Per-row: blend(0.5) + RMSNorm + exact 460th-smallest (bitonic sort of 512)
// + relu(x - kth) -> bf16.  One 512-thread block per row.
// ---------------------------------------------------------------------------
__global__ __launch_bounds__(512)
void ts_prep(const float* __restrict__ ts, const float* __restrict__ lte,
             const float* __restrict__ wgt, bf16* __restrict__ out) {
  __shared__ float red[512];
  __shared__ float srt[512];
  const int row = blockIdx.x;
  const int t   = threadIdx.x;
  const size_t base = (size_t)row * TS_;

  float x = 0.5f * ts[base + t] + 0.5f * lte[base + t];
  red[t] = x * x;
  __syncthreads();
  for (int s = 256; s > 0; s >>= 1) {
    if (t < s) red[t] += red[t + s];
    __syncthreads();
  }
  float var = red[0] * (1.0f / (float)TS_);
  float xn  = x * rsqrtf(var + 1e-6f) * wgt[t];

  srt[t] = xn;
  __syncthreads();
  for (int k = 2; k <= 512; k <<= 1)
    for (int j = k >> 1; j > 0; j >>= 1) {
      int ixj = t ^ j;
      if (ixj > t) {
        float a = srt[t], c = srt[ixj];
        if ((a > c) == ((t & k) == 0)) { srt[t] = c; srt[ixj] = a; }
      }
      __syncthreads();
    }
  float kth = srt[459];  // K_IDX-1 = int(0.9*512)-1
  float y   = xn - kth;
  out[base + t] = (bf16)(y > 0.0f ? y : 0.0f);
}

// ---------------------------------------------------------------------------
// RoPE applied to q,k (bf16 in -> bf16 out, f32 math).
// ---------------------------------------------------------------------------
__global__ void rope_kernel(const bf16* __restrict__ qi, const bf16* __restrict__ ki,
                            bf16* __restrict__ qo, bf16* __restrict__ ko) {
  int idx = blockIdx.x * blockDim.x + threadIdx.x;  // B*S*NH*64
  int i = idx & 63;
  int h = (idx >> 6) & (NH_ - 1);
  int s = (idx >> 10) & (S_ - 1);
  int b = idx >> 21;
  size_t base = ((size_t)(b * S_ + s)) * H_ + h * HD_;

  float inv = expf(-(float)(2 * i) * (9.210340371976184f / 128.0f));
  float ang = (float)s * inv;
  float sn, cs;
  sincosf(ang, &sn, &cs);

  float q1 = (float)qi[base + i], q2 = (float)qi[base + 64 + i];
  qo[base + i]      = (bf16)(q1 * cs - q2 * sn);
  qo[base + 64 + i] = (bf16)(q2 * cs + q1 * sn);
  float k1 = (float)ki[base + i], k2 = (float)ki[base + 64 + i];
  ko[base + i]      = (bf16)(k1 * cs - k2 * sn);
  ko[base + 64 + i] = (bf16)(k2 * cs + k1 * sn);
}

// ---------------------------------------------------------------------------
// Flash attention (causal). Block = 4 waves (128 thr) -> 64 q-rows; wave owns
// 16 q-rows x 128 d. K tiles of 64 keys. K and V(transposed) fragments load
// contiguously from global; only P goes through (per-wave) LDS for the
// C-layout -> A-layout conversion. No block barriers in the main loop.
// ---------------------------------------------------------------------------
__global__ __launch_bounds__(128)
void attn_kernel(const bf16* __restrict__ qb, const bf16* __restrict__ kb,
                 const bf16* __restrict__ vt, bf16* __restrict__ ob) {
  __shared__ __align__(16) bf16 PT[4 * 16 * 72];  // per-wave P: [m][key]
  const int b  = blockIdx.z, h = blockIdx.y, qt = blockIdx.x;
  const int tid  = threadIdx.x;
  const int w    = tid >> 5;
  const int ln32 = tid & 31;
  const int half = ln32 >> 4;
  const int ln   = ln32 & 15;
  const int qr0  = qt * 64 + w * 16;
  const size_t hb  = ((size_t)b * S_) * H_ + (size_t)h * HD_;
  const bf16* vtb  = vt + (size_t)b * S_ * H_ + (size_t)h * HD_ * S_;  // [d][s]

  // Q fragments (16 rows x 128 d = 4 K-chunks of 32), resident
  v16bf aq[4];
  {
    const bf16* qp = qb + hb + (size_t)(qr0 + ln) * H_;
#pragma unroll
    for (int c = 0; c < 4; ++c) {
      v8bf a0 = *reinterpret_cast<const v8bf*>(qp + c * 32 + half * 16);
      v8bf a1 = *reinterpret_cast<const v8bf*>(qp + c * 32 + 8 + half * 16);
#pragma unroll
      for (int i = 0; i < 8; ++i) { aq[c][i] = a0[i]; aq[c][8 + i] = a1[i]; }
    }
  }

  v8f o[8];
  v8f vzero = {};
#pragma unroll
  for (int d8 = 0; d8 < 8; ++d8) o[d8] = vzero;
  float m8[8], l8[8], corr8[8];
#pragma unroll
  for (int r = 0; r < 8; ++r) { m8[r] = -3.0e38f; l8[r] = 0.0f; }

  const float SCAL = 0.08838834764831845f;  // 1/sqrt(128)
  bf16* ptw = PT + w * 16 * 72;

  for (int kt = 0; kt <= qt; ++kt) {
    const int k0 = kt * 64;

    // S = Q @ K^T (16 rows x 64 keys)
    v8f sc[4];
#pragma unroll
    for (int nt = 0; nt < 4; ++nt) {
      v8f s = vzero;
      const bf16* kp = kb + hb + (size_t)(k0 + nt * 16 + ln) * H_ + half * 16;
#pragma unroll
      for (int c = 0; c < 4; ++c) {
        v8bf b0 = *reinterpret_cast<const v8bf*>(kp + c * 32);
        v8bf b1 = *reinterpret_cast<const v8bf*>(kp + c * 32 + 8);
        v16bf bk;
#pragma unroll
        for (int i = 0; i < 8; ++i) { bk[i] = b0[i]; bk[8 + i] = b1[i]; }
        s = wmma_bf16(aq[c], bk, s);
      }
      sc[nt] = s;
    }

    // scale + causal mask
#pragma unroll
    for (int nt = 0; nt < 4; ++nt) {
      int key = k0 + nt * 16 + ln;
#pragma unroll
      for (int r = 0; r < 8; ++r) {
        float v = sc[nt][r] * SCAL;
        if (key > qr0 + r + half * 8) v = -1.0e9f;
        sc[nt][r] = v;
      }
    }

    // online softmax: row max / correction (reductions stay in 16-lane halves)
#pragma unroll
    for (int r = 0; r < 8; ++r) {
      float v = fmaxf(fmaxf(sc[0][r], sc[1][r]), fmaxf(sc[2][r], sc[3][r]));
      v = fmaxf(v, __shfl_xor(v, 1, 32));
      v = fmaxf(v, __shfl_xor(v, 2, 32));
      v = fmaxf(v, __shfl_xor(v, 4, 32));
      v = fmaxf(v, __shfl_xor(v, 8, 32));
      float mn = fmaxf(m8[r], v);
      corr8[r] = __expf(m8[r] - mn);
      m8[r]    = mn;
    }
#pragma unroll
    for (int r = 0; r < 8; ++r) {
      float rs = 0.0f;
#pragma unroll
      for (int nt = 0; nt < 4; ++nt) {
        float p = __expf(sc[nt][r] - m8[r]);
        sc[nt][r] = p;
        rs += p;
      }
      rs += __shfl_xor(rs, 1, 32);
      rs += __shfl_xor(rs, 2, 32);
      rs += __shfl_xor(rs, 4, 32);
      rs += __shfl_xor(rs, 8, 32);
      l8[r] = l8[r] * corr8[r] + rs;
    }
#pragma unroll
    for (int d8 = 0; d8 < 8; ++d8)
#pragma unroll
      for (int r = 0; r < 8; ++r) o[d8][r] *= corr8[r];

    // P (C-layout) -> A-layout via per-wave LDS (in-order DS + explicit wait)
#pragma unroll
    for (int nt = 0; nt < 4; ++nt)
#pragma unroll
      for (int r = 0; r < 8; ++r)
        ptw[(r + half * 8) * 72 + nt * 16 + ln] = (bf16)sc[nt][r];
    asm volatile("s_wait_dscnt 0x0" ::: "memory");

    // O += P @ V  (V fragments contiguous from vT[d][s])
#pragma unroll
    for (int c2 = 0; c2 < 2; ++c2) {
      v16bf ap;
      const bf16* pr = ptw + ln * 72 + c2 * 32;
      v8bf p0 = *reinterpret_cast<const v8bf*>(pr + half * 16);
      v8bf p1 = *reinterpret_cast<const v8bf*>(pr + 8 + half * 16);
#pragma unroll
      for (int i = 0; i < 8; ++i) { ap[i] = p0[i]; ap[8 + i] = p1[i]; }
#pragma unroll
      for (int d8 = 0; d8 < 8; ++d8) {
        const bf16* vr = vtb + (size_t)(d8 * 16 + ln) * S_ + k0 + c2 * 32 + half * 16;
        v8bf b0 = *reinterpret_cast<const v8bf*>(vr);
        v8bf b1 = *reinterpret_cast<const v8bf*>(vr + 8);
        v16bf bv;
#pragma unroll
        for (int i = 0; i < 8; ++i) { bv[i] = b0[i]; bv[8 + i] = b1[i]; }
        o[d8] = wmma_bf16(ap, bv, o[d8]);
      }
    }
  }

  // normalize and store
#pragma unroll
  for (int r = 0; r < 8; ++r) {
    float invl = 1.0f / l8[r];
    size_t rb  = hb + (size_t)(qr0 + r + half * 8) * H_;
#pragma unroll
    for (int d8 = 0; d8 < 8; ++d8)
      ob[rb + d8 * 16 + ln] = (bf16)(o[d8][r] * invl);
  }
}

// ---------------------------------------------------------------------------
// Orchestration
// ---------------------------------------------------------------------------
extern "C" void kernel_launch(void* const* d_in, const int* in_sizes, int n_in,
                              void* d_out, int out_size, void* d_ws, size_t ws_size,
                              hipStream_t stream) {
  (void)in_sizes; (void)n_in; (void)out_size; (void)ws_size;

  const float* hs  = (const float*)d_in[0];
  const float* lte = (const float*)d_in[1];
  const float* Wq  = (const float*)d_in[2];
  const float* Wk  = (const float*)d_in[3];
  const float* Wv  = (const float*)d_in[4];
  const float* Wo  = (const float*)d_in[5];
  const float* Wts = (const float*)d_in[6];
  const float* tsw = (const float*)d_in[7];
  const float* Wor = (const float*)d_in[8];
  float* outp = (float*)d_out;

  char* ws = (char*)d_ws;
  // workspace layout (bytes), all 256-aligned; total ~180 MiB
  bf16*  hsb   = (bf16*)(ws + 0);            // 16,777,216  hidden bf16
  bf16*  WqT   = (bf16*)(ws + 16777216);     //  8,388,608  Wq^T [N][K]
  bf16*  WkT   = (bf16*)(ws + 25165824);     //  8,388,608
  bf16*  WvT   = (bf16*)(ws + 33554432);     //  8,388,608
  bf16*  WoT   = (bf16*)(ws + 41943040);     //  8,388,608
  bf16*  WtsT  = (bf16*)(ws + 50331648);     //  2,097,152  [512][2048]
  bf16*  WorT  = (bf16*)(ws + 52428800);     //  6,291,456  [4096][512]
  float* tsf   = (float*)(ws + 58720256);    //  8,388,608  ts f32
  bf16*  tfb   = (bf16*)(ws + 67108864);     //  4,194,304  ts_filtered bf16
  bf16*  qkadd = (bf16*)(ws + 71303168);     // 33,554,432  origin[:, :2H] bf16
  bf16*  qtmpb = (bf16*)(ws + 104857600);    // 16,777,216  q pre-rope (alias: attn out)
  bf16*  ktmpb = (bf16*)(ws + 121634816);    // 16,777,216  k pre-rope
  bf16*  qbb   = (bf16*)(ws + 138412032);    // 16,777,216  q post-rope
  bf16*  kbb   = (bf16*)(ws + 155189248);    // 16,777,216  k post-rope
  bf16*  vbuf  = (bf16*)(ws + 171966464);    // 16,777,216  v (natural)
  bf16*  attnb = qtmpb;                      // reuse (dead after RoPE)
  bf16*  vT    = qkadd;                      // reuse (dead after V GEMM)

  // hidden -> bf16
  cvt_f32_bf16<<<(B_ * S_ * H_ + 255) / 256, 256, 0, stream>>>(hs, hsb, B_ * S_ * H_);

  // weights -> bf16 transposed [N][K] (async global->LDS staging)
  transpose_cvt<<<dim3(H_ / 32, H_ / 32), 256, 0, stream>>>(Wq, H_, WqT, H_);
  transpose_cvt<<<dim3(H_ / 32, H_ / 32), 256, 0, stream>>>(Wk, H_, WkT, H_);
  transpose_cvt<<<dim3(H_ / 32, H_ / 32), 256, 0, stream>>>(Wv, H_, WvT, H_);
  transpose_cvt<<<dim3(H_ / 32, H_ / 32), 256, 0, stream>>>(Wo, H_, WoT, H_);
  transpose_cvt<<<dim3(TS_ / 32, H_ / 32), 256, 0, stream>>>(Wts, TS_, WtsT, H_);
  transpose_cvt<<<dim3((2 * H_) / 32, TS_ / 32), 256, 0, stream>>>(Wor, 3 * H_, WorT, TS_);

  const int M = B_ * S_;  // 4096
  dim3 blk(256);

  // ts = hs @ Wts (f32 out)
  gemm_bf16t<false, false><<<dim3(TS_ / 128, M / 128), blk, 0, stream>>>(
      hsb, H_, WtsT, H_, nullptr, 0, tsf, nullptr, TS_, H_);

  // blend + rmsnorm + kth-threshold + relu -> tfb
  ts_prep<<<M, 512, 0, stream>>>(tsf, lte, tsw, tfb);

  // origin[:, :2H] = tfb @ Worig[:, :2H] (bf16 out)
  gemm_bf16t<true, false><<<dim3((2 * H_) / 128, M / 128), blk, 0, stream>>>(
      tfb, TS_, WorT, TS_, nullptr, 0, nullptr, qkadd, 2 * H_, TS_);

  // q = hs@Wq + q_add ; k = hs@Wk + k_add ; v = hs@Wv + k_add
  gemm_bf16t<true, true><<<dim3(H_ / 128, M / 128), blk, 0, stream>>>(
      hsb, H_, WqT, H_, qkadd, 2 * H_, nullptr, qtmpb, H_, H_);
  gemm_bf16t<true, true><<<dim3(H_ / 128, M / 128), blk, 0, stream>>>(
      hsb, H_, WkT, H_, qkadd + H_, 2 * H_, nullptr, ktmpb, H_, H_);
  gemm_bf16t<true, true><<<dim3(H_ / 128, M / 128), blk, 0, stream>>>(
      hsb, H_, WvT, H_, qkadd + H_, 2 * H_, nullptr, vbuf, H_, H_);

  // RoPE on q,k
  rope_kernel<<<(B_ * S_ * NH_ * 64) / 256, 256, 0, stream>>>(qtmpb, ktmpb, qbb, kbb);

  // V -> vT[b][h*HD+d][s] (per-batch transpose, async staging)
  transpose_b16<<<dim3(H_ / 32, S_ / 32, B_), 256, 0, stream>>>(vbuf, vT);

  // flash attention (causal)
  attn_kernel<<<dim3(S_ / 64, NH_, B_), 128, 0, stream>>>(qbb, kbb, vT, attnb);

  // out = attn @ Wo (f32 -> d_out)
  gemm_bf16t<false, false><<<dim3(H_ / 128, M / 128), blk, 0, stream>>>(
      attnb, H_, WoT, H_, nullptr, 0, outp, nullptr, H_, H_);
}
